// KANConvNet_51238959841378
// MI455X (gfx1250) — compile-verified, matrix operand order checked by reading, not run on hardware
//
#include <hip/hip_runtime.h>
#include <hip/hip_bf16.h>

typedef __attribute__((ext_vector_type(16))) _Float16 v16h;
typedef __attribute__((ext_vector_type(8)))  _Float16 v8h;
typedef __attribute__((ext_vector_type(8)))  float    v8f;
typedef __attribute__((ext_vector_type(4)))  int      v4i;

#define KAN_MAXK 5184   // stage3: 64*3*3*9

// ---------------------------------------------------------------------------
// B-spline basis (grid=5, order=3, range [-1,1]) + SiLU, per scalar input.
// Cox-de Boor fully unrolled; grid points are compile-time constants.
// ---------------------------------------------------------------------------
__device__ __forceinline__ void kan_basis9(float x, float* out9) {
  const float h = 0.4f;                 // 2/GRID_SIZE
  float gr[12];
#pragma unroll
  for (int j = 0; j < 12; ++j) gr[j] = -1.0f + (float)(j - 3) * h;
  float b[11];
#pragma unroll
  for (int j = 0; j < 11; ++j) b[j] = (x >= gr[j] && x < gr[j + 1]) ? 1.0f : 0.0f;
#pragma unroll
  for (int k = 1; k <= 3; ++k) {
#pragma unroll
    for (int j = 0; j < 10; ++j) {      // ascending j: b[j+1] still holds old value
      if (j < 11 - k) {
        b[j] = (x - gr[j]) / (gr[j + k] - gr[j]) * b[j]
             + (gr[j + k + 1] - x) / (gr[j + k + 1] - gr[j + 1]) * b[j + 1];
      }
    }
  }
  float sig = 1.0f / (1.0f + __expf(-x));
  out9[0] = x * sig;                    // SiLU base branch
#pragma unroll
  for (int f = 0; f < 8; ++f) out9[1 + f] = b[f];
}

// ---------------------------------------------------------------------------
// Per-pixel featurization: in (fp32 pixel stream) -> feat[pixel*9 + {0..8}] f16
// Pixel-level (not patch-level) so overlapping conv windows reuse it.
// ---------------------------------------------------------------------------
__global__ void kan_featurize(const float* __restrict__ in,
                              _Float16* __restrict__ feat, int n) {
  int i = blockIdx.x * blockDim.x + threadIdx.x;
  if (i >= n) return;
  float v9[9];
  kan_basis9(in[i], v9);
  _Float16* o = feat + (size_t)i * 9;
#pragma unroll
  for (int j = 0; j < 9; ++j) o[j] = (_Float16)v9[j];
}

// ---------------------------------------------------------------------------
// Pack combined weight  Wp[o][i*9+0]=bw[o,i], Wp[o][i*9+1+f]=sw[o,i,f]*ss[o,i]
// into (N, K) row-major f16, so B-fragments load as two b128 per lane.
// ---------------------------------------------------------------------------
__global__ void kan_pack_w(const float* __restrict__ bw,
                           const float* __restrict__ sw,
                           const float* __restrict__ ss,
                           _Float16* __restrict__ wp, int nout, int nin) {
  int t = blockIdx.x * blockDim.x + threadIdx.x;
  if (t >= nout * nin) return;
  _Float16* dst = wp + (size_t)t * 9;
  dst[0] = (_Float16)bw[t];
  float s = ss[t];
#pragma unroll
  for (int f = 0; f < 8; ++f) dst[1 + f] = (_Float16)(sw[(size_t)t * 8 + f] * s);
}

// ---------------------------------------------------------------------------
// Fused WMMA GEMM (M,K)x(K,N) + LayerNorm over channels + ReLU.
//   * wave = 32-row M tile (two 16-row A fragments), all NT n-tiles in v8f regs
//     -> each B fragment load feeds TWO v_wmma ops
//   * A gathered from pixel-level feat via LDS offset LUT: rowBase(m)+ktab[k];
//     LUT read as int4 (ds_load_b128)
//   * B loaded contiguously (2x b128/lane) from packed (N,K) f16 weights
//   * epilogue: 32xN tile in LDS, one LayerNorm row per lane, gamma/beta, ReLU,
//     written in (B,Cout,Ho,Wo) layout for the next stage's featurize pass
// ---------------------------------------------------------------------------
template <int NT>
__global__ __launch_bounds__(64)
void kan_gemm_ln_relu(const _Float16* __restrict__ feat,
                      const _Float16* __restrict__ wp,
                      const float* __restrict__ gma,
                      const float* __restrict__ bta,
                      float* __restrict__ out,
                      int K, int Cin, int Hin, int Win,
                      int Ho, int Wo, int kh, int kw, int stride) {
  __shared__ int   ktab[KAN_MAXK];
  __shared__ float lnt[2][32][NT * 16];

  const int tid  = threadIdx.x;
  const int lane = tid & 31;
  const int wave = tid >> 5;
  const int N    = NT * 16;

  // k -> feat element-offset LUT (separable from output position)
  for (int k = tid; k < K; k += 64) {
    int i = k / 9, comp = k % 9;
    int c   = i / (kh * kw);
    int rem = i % (kh * kw);
    int ky = rem / kw, kx = rem % kw;
    ktab[k] = ((c * Hin + ky) * Win + kx) * 9 + comp;
  }
  __syncthreads();

  const int wtile = blockIdx.x * 2 + wave;   // 32 rows per wave
  const int hw    = Ho * Wo;
  const int r     = lane & 15;

  int rowBase[2];
#pragma unroll
  for (int f = 0; f < 2; ++f) {
    int m    = wtile * 32 + f * 16 + r;
    int bidx = m / hw, rem2 = m % hw;
    int oy = rem2 / Wo, ox = rem2 % Wo;
    rowBase[f] = (((bidx * Cin) * Hin + oy * stride) * Win + ox * stride) * 9;
  }

  v8f acc[2][NT];
#pragma unroll
  for (int f = 0; f < 2; ++f)
#pragma unroll
    for (int t = 0; t < NT; ++t)
#pragma unroll
      for (int q = 0; q < 8; ++q) acc[f][t][q] = 0.0f;

  const int khalf = (lane >> 4) << 3;        // lanes 0-15: +0, lanes 16-31: +8
  const _Float16* wrow[NT];
#pragma unroll
  for (int t = 0; t < NT; ++t)
    wrow[t] = wp + (size_t)(t * 16 + r) * K;

  const int kchunks = K >> 5;
  for (int kt = 0; kt < kchunks; ++kt) {
    const int kb = kt * 32 + khalf;          // multiple of 8 -> 16B-aligned LUT reads
    v4i o0 = *(const v4i*)&ktab[kb];
    v4i o1 = *(const v4i*)&ktab[kb + 4];
    v4i o2 = *(const v4i*)&ktab[kb + 16];
    v4i o3 = *(const v4i*)&ktab[kb + 20];
    int offs[16];
#pragma unroll
    for (int e = 0; e < 4; ++e) {
      offs[e]      = o0[e];
      offs[4 + e]  = o1[e];
      offs[8 + e]  = o2[e];
      offs[12 + e] = o3[e];
    }
    // A fragments: 16x32 f16, ISA per-lane layout (e<8 -> kb+e, e>=8 -> kb+16+(e-8))
    v16h a[2];
#pragma unroll
    for (int f = 0; f < 2; ++f)
#pragma unroll
      for (int e = 0; e < 8; ++e) {
        a[f][e]     = feat[rowBase[f] + offs[e]];
        a[f][e + 8] = feat[rowBase[f] + offs[8 + e]];
      }
#pragma unroll
    for (int t = 0; t < NT; ++t) {
      // B fragment: 32x16 f16, same K striping, n = lane&15 within tile
      v8h lo = *(const v8h*)(wrow[t] + kb);
      v8h hi = *(const v8h*)(wrow[t] + kb + 16);
      v16h bf;
#pragma unroll
      for (int e = 0; e < 8; ++e) { bf[e] = lo[e]; bf[e + 8] = hi[e]; }
      acc[0][t] = __builtin_amdgcn_wmma_f32_16x16x32_f16(
          false, a[0], false, bf, (short)0, acc[0][t], false, false);
      acc[1][t] = __builtin_amdgcn_wmma_f32_16x16x32_f16(
          false, a[1], false, bf, (short)0, acc[1][t], false, false);
    }
  }

  // ---- epilogue: LayerNorm over channels + ReLU --------------------------
  const int rr0 = (lane < 16) ? 0 : 8;       // C/D layout: VGPR q -> row q(+8)
  const int n0  = lane & 15;
#pragma unroll
  for (int f = 0; f < 2; ++f)
#pragma unroll
    for (int t = 0; t < NT; ++t)
#pragma unroll
      for (int q = 0; q < 8; ++q)
        lnt[wave][f * 16 + q + rr0][t * 16 + n0] = acc[f][t][q];
  __syncthreads();

  {
    // one output row (= one spatial position) per lane
    int m    = wtile * 32 + lane;
    int bidx = m / hw, rem2 = m % hw;
    int oy = rem2 / Wo, ox = rem2 % Wo;
    float sum = 0.f, sq = 0.f;
#pragma unroll
    for (int n = 0; n < NT * 16; ++n) {
      float v = lnt[wave][lane][n];
      sum += v; sq += v * v;
    }
    float mu   = sum / (float)N;
    float var  = sq / (float)N - mu * mu;    // biased, matches jnp.var
    float rstd = rsqrtf(var + 1e-5f);
    for (int n = 0; n < NT * 16; ++n) {
      float v = (lnt[wave][lane][n] - mu) * rstd * gma[n] + bta[n];
      out[((bidx * N + n) * Ho + oy) * Wo + ox] = fmaxf(v, 0.0f);
    }
  }
}

// ---------------------------------------------------------------------------
extern "C" void kernel_launch(void* const* d_in, const int* in_sizes, int n_in,
                              void* d_out, int out_size, void* d_ws, size_t ws_size,
                              hipStream_t stream) {
  const float* x   = (const float*)d_in[0];
  const float* bw1 = (const float*)d_in[1];
  const float* sw1 = (const float*)d_in[2];
  const float* ss1 = (const float*)d_in[3];
  const float* g1  = (const float*)d_in[4];
  const float* be1 = (const float*)d_in[5];
  const float* bw2 = (const float*)d_in[6];
  const float* sw2 = (const float*)d_in[7];
  const float* ss2 = (const float*)d_in[8];
  const float* g2  = (const float*)d_in[9];
  const float* be2 = (const float*)d_in[10];
  const float* bw3 = (const float*)d_in[11];
  const float* sw3 = (const float*)d_in[12];
  const float* ss3 = (const float*)d_in[13];
  const float* g3  = (const float*)d_in[14];
  const float* be3 = (const float*)d_in[15];

  char* ws = (char*)d_ws;
  // workspace layout (all 256B aligned):
  //   y1:   32*32*52*38 f32  =  8,093,696 B
  //   y2:   32*64*25*18 f32  =  3,686,400 B
  //   feat: max 2,023,424 px * 9 f16 = 36,421,632 B
  //   wp:   max 64*5184 f16  =    663,552 B     (total ~48.9 MB)
  float*    y1   = (float*)ws;
  float*    y2   = (float*)(ws + 8093696);
  _Float16* feat = (_Float16*)(ws + 8093696 + 3686400);
  _Float16* wp   = (_Float16*)(ws + 8093696 + 3686400 + 36421632);
  float*    y3   = (float*)d_out;

  const int TB = 256;
  // ---------------- stage 1: 1x212x156 --(k8,s4)--> 32x52x38 ----------------
  {
    int npix = 32 * 1 * 212 * 156;            // 1,058,304
    kan_pack_w<<<(32 * 64 + TB - 1) / TB, TB, 0, stream>>>(bw1, sw1, ss1, wp, 32, 64);
    kan_featurize<<<(npix + TB - 1) / TB, TB, 0, stream>>>(x, feat, npix);
    int M = 32 * 52 * 38;                     // 63232, /64 = 988 blocks
    kan_gemm_ln_relu<2><<<M / 64, 64, 0, stream>>>(
        feat, wp, g1, be1, y1,
        /*K=*/64 * 9, /*Cin=*/1, /*Hin=*/212, /*Win=*/156,
        /*Ho=*/52, /*Wo=*/38, /*kh=*/8, /*kw=*/8, /*stride=*/4);
  }
  // ---------------- stage 2: 32x52x38 --(k4,s2)--> 64x25x18 -----------------
  {
    int npix = 32 * 32 * 52 * 38;             // 2,023,424
    kan_pack_w<<<(64 * 512 + TB - 1) / TB, TB, 0, stream>>>(bw2, sw2, ss2, wp, 64, 512);
    kan_featurize<<<(npix + TB - 1) / TB, TB, 0, stream>>>(y1, feat, npix);
    int M = 32 * 25 * 18;                     // 14400, /64 = 225 blocks
    kan_gemm_ln_relu<4><<<M / 64, 64, 0, stream>>>(
        feat, wp, g2, be2, y2,
        /*K=*/512 * 9, /*Cin=*/32, /*Hin=*/52, /*Win=*/38,
        /*Ho=*/25, /*Wo=*/18, /*kh=*/4, /*kw=*/4, /*stride=*/2);
  }
  // ---------------- stage 3: 64x25x18 --(k3,s1)--> 64x23x16 (output) --------
  {
    int npix = 32 * 64 * 25 * 18;             // 921,600
    kan_pack_w<<<(64 * 576 + TB - 1) / TB, TB, 0, stream>>>(bw3, sw3, ss3, wp, 64, 576);
    kan_featurize<<<(npix + TB - 1) / TB, TB, 0, stream>>>(y2, feat, npix);
    int M = 32 * 23 * 16;                     // 11776, /64 = 184 blocks
    kan_gemm_ln_relu<4><<<M / 64, 64, 0, stream>>>(
        feat, wp, g3, be3, y3,
        /*K=*/576 * 9, /*Cin=*/64, /*Hin=*/25, /*Win=*/18,
        /*Ho=*/23, /*Wo=*/16, /*kh=*/3, /*kw=*/3, /*stride=*/1);
  }
}
